// FMoE_87016037417562
// MI455X (gfx1250) — matrix operand order repaired and checked by reading, not compiled, not run on hardware
//
#include <hip/hip_runtime.h>
#include <hip/hip_bf16.h>

// ---------------- problem constants (match reference) ----------------
constexpr int E = 8;      // experts
constexpr int D = 768;    // d_model
constexpr int H = 3072;   // expert hidden
constexpr int K = 2;      // top-k
constexpr int N = 4096;   // tokens
constexpr int C = 2048;   // per-expert capacity

typedef __attribute__((ext_vector_type(16))) __bf16 v16bf;
typedef __attribute__((ext_vector_type(8)))  float  v8f;

// ---------------------------------------------------------------------
// small helpers
// ---------------------------------------------------------------------
__device__ __forceinline__ float gelu_tanh(float v) {
    // jax.nn.gelu default (approximate=True)
    float c = 0.7978845608028654f * (v + 0.044715f * v * v * v);
    return 0.5f * v * (1.0f + tanhf(c));
}

// ---------------------------------------------------------------------
// 0) zero the per-expert slot counters (must be re-zeroed every launch)
// ---------------------------------------------------------------------
__global__ void zero_cnt_kernel(int* __restrict__ cnt) {
    if (threadIdx.x < E) cnt[threadIdx.x] = 0;
}

// ---------------------------------------------------------------------
// 1) gating: logits = x @ gate_w + gate_b, top-2, softmax over the 2
//    one block (128 threads) per token
// ---------------------------------------------------------------------
__global__ __launch_bounds__(128) void gate_kernel(
    const float* __restrict__ x,      // [N,D]
    const float* __restrict__ gw,     // [D,E]
    const float* __restrict__ gb,     // [E]
    int*   __restrict__ top_i,        // [N,K]
    float* __restrict__ gate_s)       // [N,K]
{
    const int n   = blockIdx.x;
    const int tid = threadIdx.x;
    float acc[E];
#pragma unroll
    for (int e = 0; e < E; ++e) acc[e] = 0.0f;

    const float* xr = x + (size_t)n * D;
    for (int d = tid; d < D; d += 128) {
        const float xv = xr[d];
        const float* wr = gw + (size_t)d * E;   // 8 contiguous floats
#pragma unroll
        for (int e = 0; e < E; ++e) acc[e] += xv * wr[e];
    }

    __shared__ float sm[128][E];
#pragma unroll
    for (int e = 0; e < E; ++e) sm[tid][e] = acc[e];
    __syncthreads();
    for (int s = 64; s > 0; s >>= 1) {
        if (tid < s) {
#pragma unroll
            for (int e = 0; e < E; ++e) sm[tid][e] += sm[tid + s][e];
        }
        __syncthreads();
    }

    if (tid == 0) {
        float lg[E];
#pragma unroll
        for (int e = 0; e < E; ++e) lg[e] = sm[0][e] + gb[e];
        // top-2, ties -> lower index first (matches jax.lax.top_k)
        int i0 = 0;
#pragma unroll
        for (int e = 1; e < E; ++e) if (lg[e] > lg[i0]) i0 = e;
        int i1 = -1;
#pragma unroll
        for (int e = 0; e < E; ++e) {
            if (e == i0) continue;
            if (i1 < 0 || lg[e] > lg[i1]) i1 = e;
        }
        const float m  = fmaxf(lg[i0], lg[i1]);
        const float e0 = __expf(lg[i0] - m);
        const float e1 = __expf(lg[i1] - m);
        const float inv = 1.0f / (e0 + e1);
        top_i[n * K + 0] = i0;
        top_i[n * K + 1] = i1;
        gate_s[n * K + 0] = e0 * inv;
        gate_s[n * K + 1] = e1 * inv;
    }
}

// ---------------------------------------------------------------------
// 2) fp32 -> bf16 weight conversion (bf16 weights fit in the 192MB L2)
// ---------------------------------------------------------------------
__global__ __launch_bounds__(256) void cvt_f32_bf16_kernel(
    const float* __restrict__ in, __bf16* __restrict__ out, long n)
{
    long i = ((long)blockIdx.x * 256 + threadIdx.x) * 4;
    if (i + 3 < n) {
        const float4 f = *(const float4*)(in + i);
        union { __bf16 o[4]; uint2 u; } u;
        u.o[0] = (__bf16)f.x; u.o[1] = (__bf16)f.y;
        u.o[2] = (__bf16)f.z; u.o[3] = (__bf16)f.w;
        *(uint2*)(out + i) = u.u;
    }
}

// ---------------------------------------------------------------------
// 3) scatter tokens into expert capacity buffers (bf16), record pos
//    one block (192 threads) per (token,k) slot; 4 floats per thread
// ---------------------------------------------------------------------
__global__ __launch_bounds__(192) void scatter_kernel(
    const float* __restrict__ x,     // [N,D]
    const int*   __restrict__ top_i, // [N*K]
    int*   __restrict__ cnt,         // [E]
    int*   __restrict__ pos,         // [N*K]
    __bf16* __restrict__ xe)         // [E,C,D]
{
    const int s = blockIdx.x;        // slot id
    const int n = s >> 1;            // K == 2
    __shared__ int sp, se;
    if (threadIdx.x == 0) {
        const int e = top_i[s];
        const int p = atomicAdd(&cnt[e], 1);
        pos[s] = (p < C) ? p : -1;
        sp = p; se = e;
    }
    __syncthreads();
    const int p = sp;
    if (p >= C) return;              // capacity drop
    const int e = se;

    const float4 f = *(const float4*)(x + (size_t)n * D + threadIdx.x * 4);
    union { __bf16 o[4]; uint2 u; } u;
    u.o[0] = (__bf16)f.x; u.o[1] = (__bf16)f.y;
    u.o[2] = (__bf16)f.z; u.o[3] = (__bf16)f.w;
    *(uint2*)(xe + ((size_t)e * C + p) * D + threadIdx.x * 4) = u.u;
}

// ---------------------------------------------------------------------
// 4) grouped expert GEMM with WMMA bf16 -> f32
//    block: 256 thr = 8 waves; block tile 64(M) x 128(N); wave 32x32;
//    K staged through LDS in steps of 32; early-exit on expert count.
// ---------------------------------------------------------------------
template<bool GELU, typename OutT>
__global__ __launch_bounds__(256) void moe_gemm_kernel(
    const __bf16* __restrict__ A,    // [E,C,Kd]
    const __bf16* __restrict__ Bw,   // [E,Kd,Nd]
    const float*  __restrict__ bias, // [E,Nd]
    OutT*         __restrict__ Cout, // [E,C,Nd]
    const int*    __restrict__ cnt,  // [E]
    int Kd, int Nd)
{
    const int e = blockIdx.z;
    int count = cnt[e]; if (count > C) count = C;
    const int mBase = blockIdx.y * 64;
    if (mBase >= count) return;      // skip unoccupied capacity tiles
    const int nBase = blockIdx.x * 128;

    __shared__ alignas(16) __bf16 As[64][40];   // [m][k], row = 80B (16B aligned)
    __shared__ alignas(16) __bf16 Bs[128][40];  // transposed: [n][k]

    const int tid  = threadIdx.x;
    const int lane = tid & 31;
    const int wave = tid >> 5;
    const int wm = (wave >> 2) * 32;             // 0 / 32
    const int wn = (wave & 3) * 32;              // 0 / 32 / 64 / 96
    const int half = lane >> 4;                  // 0 / 1
    const int r16  = lane & 15;

    v8f acc[2][2];
#pragma unroll
    for (int i = 0; i < 2; ++i)
#pragma unroll
        for (int j = 0; j < 2; ++j) acc[i][j] = (v8f)(0.0f);

    const __bf16* Ae = A  + (size_t)e * C  * Kd;
    const __bf16* Be = Bw + (size_t)e * Kd * Nd;

    for (int kb = 0; kb < Kd; kb += 32) {
        // stage A: 64 rows x 32 bf16; one uint4 (8 bf16) per thread
        {
            const int r = tid >> 2, cc = (tid & 3) * 8;
            const uint4 v = *(const uint4*)(Ae + (size_t)(mBase + r) * Kd + kb + cc);
            *(uint4*)(&As[r][cc]) = v;
        }
        // stage B transposed: 32 rows x 128 bf16; 16 bf16 per thread
        {
            const int k = tid >> 3, n0 = (tid & 7) * 16;
            const __bf16* src = Be + (size_t)(kb + k) * Nd + nBase + n0;
            union { __bf16 t[16]; uint4 q[2]; } u;
            u.q[0] = *(const uint4*)(src);
            u.q[1] = *(const uint4*)(src + 8);
#pragma unroll
            for (int j = 0; j < 16; ++j) Bs[n0 + j][k] = u.t[j];
        }
        __syncthreads();

        // A fragment: lanes 0-15 -> K 0-7 & 16-23, lanes 16-31 -> K 8-15 & 24-31
        v16bf afrag[2], bfrag[2];
#pragma unroll
        for (int i = 0; i < 2; ++i) {
            const __bf16* row = &As[wm + i * 16 + r16][0];
            union { uint4 q[2]; v16bf v; } u;
            u.q[0] = *(const uint4*)(row + half * 8);
            u.q[1] = *(const uint4*)(row + 16 + half * 8);
            afrag[i] = u.v;
        }
        // B fragment: lane = column n, lanes 0-15 -> K 0-7 & 16-23, lanes 16-31 -> K 8-15 & 24-31
#pragma unroll
        for (int i = 0; i < 2; ++i) {
            const __bf16* row = &Bs[wn + i * 16 + r16][0];
            union { uint4 q[2]; v16bf v; } u;
            u.q[0] = *(const uint4*)(row + half * 8);
            u.q[1] = *(const uint4*)(row + 16 + half * 8);
            bfrag[i] = u.v;
        }
#pragma unroll
        for (int mi = 0; mi < 2; ++mi)
#pragma unroll
            for (int ni = 0; ni < 2; ++ni)
                acc[mi][ni] = __builtin_amdgcn_wmma_f32_16x16x32_bf16(
                    false, afrag[mi], false, bfrag[ni],
                    (short)0, acc[mi][ni], false, false);
        __syncthreads();
    }

    // epilogue: +bias, optional GELU, store
    OutT* Ce = Cout + (size_t)e * C * Nd;
    const float* be = bias + (size_t)e * Nd;
#pragma unroll
    for (int mi = 0; mi < 2; ++mi) {
#pragma unroll
        for (int ni = 0; ni < 2; ++ni) {
            const int gn = nBase + wn + ni * 16 + r16;
            const float bv = be[gn];
#pragma unroll
            for (int i = 0; i < 8; ++i) {
                const int gm = mBase + wm + mi * 16 + half * 8 + i;
                float v = acc[mi][ni][i] + bv;
                if (GELU) v = gelu_tanh(v);
                Ce[(size_t)gm * Nd + gn] = (OutT)v;
            }
        }
    }
}

// ---------------------------------------------------------------------
// 5) gather back to token order, combine with gate scores
// ---------------------------------------------------------------------
__global__ __launch_bounds__(256) void gather_kernel(
    const float* __restrict__ ye,     // [E,C,D]
    const int*   __restrict__ top_i,  // [N*K]
    const int*   __restrict__ pos,    // [N*K]
    const float* __restrict__ gate_s, // [N*K]
    float* __restrict__ out)          // [N,D]
{
    const int n = blockIdx.x;
    const int e0 = top_i[n * K + 0], e1 = top_i[n * K + 1];
    const int p0 = pos[n * K + 0],   p1 = pos[n * K + 1];
    const float s0 = gate_s[n * K + 0], s1 = gate_s[n * K + 1];
    const float* r0 = (p0 >= 0) ? ye + ((size_t)e0 * C + p0) * D : nullptr;
    const float* r1 = (p1 >= 0) ? ye + ((size_t)e1 * C + p1) * D : nullptr;
    for (int d = threadIdx.x; d < D; d += 256) {
        float v = 0.0f;
        if (r0) v += s0 * r0[d];
        if (r1) v += s1 * r1[d];
        out[(size_t)n * D + d] = v;
    }
}

// ---------------------------------------------------------------------
// launcher
// ---------------------------------------------------------------------
extern "C" void kernel_launch(void* const* d_in, const int* in_sizes, int n_in,
                              void* d_out, int out_size, void* d_ws, size_t ws_size,
                              hipStream_t stream)
{
    (void)in_sizes; (void)n_in; (void)out_size; (void)ws_size;

    const float* moe_inp = (const float*)d_in[0];   // [N,D]
    const float* gate_w  = (const float*)d_in[1];   // [D,E]
    const float* gate_b  = (const float*)d_in[2];   // [E]
    const float* w1      = (const float*)d_in[3];   // [E,D,H]
    const float* b1      = (const float*)d_in[4];   // [E,H]
    const float* w2      = (const float*)d_in[5];   // [E,H,D]
    const float* b2      = (const float*)d_in[6];   // [E,D]
    float* out = (float*)d_out;                     // [N,D]

    // workspace carve-out (256B aligned)
    char* ws = (char*)d_ws;
    size_t off = 0;
    auto carve = [&](size_t bytes) -> char* {
        char* p = ws + off;
        off = (off + bytes + 255) & ~(size_t)255;
        return p;
    };
    int*    cnt    = (int*)   carve(E * sizeof(int));
    int*    top_i  = (int*)   carve((size_t)N * K * sizeof(int));
    int*    pos    = (int*)   carve((size_t)N * K * sizeof(int));
    float*  gate_s = (float*) carve((size_t)N * K * sizeof(float));
    __bf16* w1b    = (__bf16*)carve((size_t)E * D * H * sizeof(__bf16));
    __bf16* w2b    = (__bf16*)carve((size_t)E * H * D * sizeof(__bf16));
    __bf16* xe     = (__bf16*)carve((size_t)E * C * D * sizeof(__bf16));
    __bf16* h      = (__bf16*)carve((size_t)E * C * H * sizeof(__bf16));
    float*  ye     = (float*) carve((size_t)E * C * D * sizeof(float));

    // 0) counters
    zero_cnt_kernel<<<1, 32, 0, stream>>>(cnt);

    // 1) gating
    gate_kernel<<<N, 128, 0, stream>>>(moe_inp, gate_w, gate_b, top_i, gate_s);

    // 2) weights -> bf16 (75MB total; resident in the 192MB L2 afterwards)
    {
        const long nw = (long)E * D * H;                 // same for w1 and w2
        const int blocks = (int)((nw + 256L * 4 - 1) / (256L * 4));
        cvt_f32_bf16_kernel<<<blocks, 256, 0, stream>>>(w1, w1b, nw);
        cvt_f32_bf16_kernel<<<blocks, 256, 0, stream>>>(w2, w2b, nw);
    }

    // 3) scatter (D/4 = 192 threads: one float4 per thread)
    scatter_kernel<<<N * K, 192, 0, stream>>>(moe_inp, top_i, cnt, pos, xe);

    // 4) expert FFN: GEMM1 (+b1, GELU, bf16 out) then GEMM2 (+b2, f32 out)
    moe_gemm_kernel<true, __bf16><<<dim3(H / 128, C / 64, E), 256, 0, stream>>>(
        xe, w1b, b1, h, cnt, D, H);
    moe_gemm_kernel<false, float><<<dim3(D / 128, C / 64, E), 256, 0, stream>>>(
        h, w2b, b2, ye, cnt, H, D);

    // 5) gather + combine
    gather_kernel<<<N, 256, 0, stream>>>(ye, top_i, pos, gate_s, out);
}
